// EncoderLayer_4535485465040
// MI455X (gfx1250) — matrix operand order repaired and checked by reading, not compiled, not run on hardware
//
#include <hip/hip_runtime.h>
#include <hip/hip_bf16.h>
#include <math.h>

// ---------------------------------------------------------------------------
// Transformer encoder layer for gfx1250 (MI455X), wave32 + WMMA f16 (f32 acc)
// Fused attention (scores+mask+softmax+context in LDS), 64x64-per-wave GEMMs.
// ---------------------------------------------------------------------------

typedef __attribute__((ext_vector_type(16))) _Float16 v16h;
typedef __attribute__((ext_vector_type(8)))  float    v8f;
typedef __attribute__((ext_vector_type(4)))  int      v4i;
typedef _Float16 h16;

#define D_MODEL 768
#define NH      12
#define DK      64
#define DFF     3072
#define BB      8
#define SS      1024
#define MTOT    (BB * SS)        // 8192 token rows
#define SCP     1028             // padded LDS row stride (floats) for scores

#if defined(__has_builtin)
#if __has_builtin(__builtin_amdgcn_global_load_async_to_lds_b128)
#define ASYNC_OK 1
typedef __attribute__((address_space(1))) v4i* gv4i_p;
typedef __attribute__((address_space(3))) v4i* lv4i_p;
#endif
#endif

// ---------------- WMMA core ----------------
__device__ __forceinline__ v8f wmma_f16(v16h a, v16h b, v8f c) {
  return __builtin_amdgcn_wmma_f32_16x16x32_f16(
      /*neg_a=*/false, a, /*neg_b=*/false, b,
      /*c_mod=*/(short)0, c, /*reuse_a=*/false, /*reuse_b=*/false);
}

// A fragment (16x32, row-major A). lane l(0..15),hh: halves0..7 = K k0+hh*8..,
// halves8..15 = +16. Two contiguous 16B loads per lane.
__device__ __forceinline__ v16h frag_a_h(const h16* A, int lda, int m0, int k0,
                                         int l, int hh) {
  const h16* p = A + (size_t)(m0 + l) * lda + k0 + hh * 8;
  union { float4 f[2]; v16h v; } u;
  u.f[0] = *reinterpret_cast<const float4*>(p);
  u.f[1] = *reinterpret_cast<const float4*>(p + 16);
  return u.v;
}

// B fragment (32x16) from Bt stored N-major: lane l = column n0+l,
// halves0..7 = K k0+hh*8.., halves8..15 = +16. Two 16B loads per lane.
__device__ __forceinline__ v16h frag_b_h(const h16* Bt, int ldb, int n0, int k0,
                                         int l, int hh) {
  const h16* p = Bt + (size_t)(n0 + l) * ldb + k0 + hh * 8;
  union { float4 f[2]; v16h v; } u;
  u.f[0] = *reinterpret_cast<const float4*>(p);
  u.f[1] = *reinterpret_cast<const float4*>(p + 16);
  return u.v;
}

// A fragment sourced from f32 (attention probabilities in LDS), cvt to f16.
__device__ __forceinline__ v16h frag_a_f32(const float* A, int lda, int m0, int k0,
                                           int l, int hh) {
  const float* p = A + (size_t)(m0 + l) * lda + k0 + hh * 8;
  v16h r;
#pragma unroll
  for (int i = 0; i < 8; ++i) r[i] = (_Float16)p[i];
#pragma unroll
  for (int i = 0; i < 8; ++i) r[8 + i] = (_Float16)p[16 + i];
  return r;
}

// ---------------- converters ----------------
__global__ void cvt_f32_to_f16(const float* __restrict__ src, h16* __restrict__ dst,
                               int n) {
  int i = blockIdx.x * blockDim.x + threadIdx.x;
  if (i < n) dst[i] = (h16)src[i];
}

__global__ void cvt_transpose_h(const float* __restrict__ src, h16* __restrict__ dst,
                                int K, int N) {
  int i = blockIdx.x * blockDim.x + threadIdx.x;
  if (i < K * N) {
    int k = i / N, n = i % N;
    dst[(size_t)n * K + k] = (h16)src[i];
  }
}

// ---------------- generic WMMA GEMM, 128x128 per WG (64x64 per wave) ---------
enum { EPI_F32 = 0, EPI_GELU_F16 = 1, EPI_QK = 2, EPI_VT = 3 };

__global__ void __launch_bounds__(128) gemm_ws(
    const h16* __restrict__ A, const h16* __restrict__ Bt,
    const float* __restrict__ bias, int K, int lda, int ldb,
    float* __restrict__ Cf, h16* __restrict__ Ch, int ldc,
    int mode, float scale) {
  const int lane = threadIdx.x & 31, l = lane & 15, hh = lane >> 4;
  const int wave = threadIdx.x >> 5;
  const int m0 = blockIdx.y * 128 + (wave >> 1) * 64;
  const int n0 = blockIdx.x * 128 + (wave & 1) * 64;

  // lane-row base pointers for prefetch
  const h16* apf = A + (size_t)(m0 + l) * lda;
  const h16* bpf = Bt + (size_t)(n0 + l) * ldb;

  v8f acc[4][4] = {};
  for (int k0 = 0; k0 < K; k0 += 32) {
    __builtin_prefetch((const void*)(apf + k0 + 128), 0, 1);  // global_prefetch_b8
    __builtin_prefetch((const void*)(bpf + k0 + 128), 0, 1);
    v16h a[4], b[4];
#pragma unroll
    for (int i = 0; i < 4; ++i) a[i] = frag_a_h(A, lda, m0 + i * 16, k0, l, hh);
#pragma unroll
    for (int j = 0; j < 4; ++j) b[j] = frag_b_h(Bt, ldb, n0 + j * 16, k0, l, hh);
#pragma unroll
    for (int i = 0; i < 4; ++i)
#pragma unroll
      for (int j = 0; j < 4; ++j)
        acc[i][j] = wmma_f16(a[i], b[j], acc[i][j]);
  }

#pragma unroll
  for (int ti = 0; ti < 4; ++ti)
#pragma unroll
    for (int tj = 0; tj < 4; ++tj)
#pragma unroll
      for (int r = 0; r < 8; ++r) {
        const int m = m0 + ti * 16 + r + hh * 8;
        const int n = n0 + tj * 16 + l;
        float v = acc[ti][tj][r] + bias[n];
        if (mode == EPI_F32) {
          Cf[(size_t)m * ldc + n] = v;
        } else if (mode == EPI_GELU_F16) {
          float g = 0.5f * v * (1.0f + erff(v * 0.70710678118654752f));
          Ch[(size_t)m * ldc + n] = (h16)g;
        } else if (mode == EPI_QK) {       // -> [b, h, s, d] f16 (scaled)
          int b = m >> 10, s = m & 1023, hid = n >> 6, d = n & 63;
          Ch[(((size_t)b * NH + hid) * SS + s) * DK + d] = (h16)(v * scale);
        } else {                           // EPI_VT -> [b, h, d, s] f16
          int b = m >> 10, s = m & 1023, hid = n >> 6, d = n & 63;
          Ch[(((size_t)b * NH + hid) * DK + d) * SS + s] = (h16)v;
        }
      }
}

// ---------------- fused attention ----------------
// One WG (128 thr, 4 waves) per (b*NH+h, q-block of 32).
//  phase 0: stage Q tile (32x64 f16) to LDS (async copy if available)
//  phase 1: scores 32x1024 = Q @ K^T into LDS (f32, padded stride)
//  phase 2: mask + softmax in LDS; coalesced f32 prob write to d_out
//  phase 3: context 32x64 = probs(LDS) @ V^T(global), f16 -> ctx
__global__ void __launch_bounds__(128) fused_attn(
    const h16* __restrict__ Qh, const h16* __restrict__ Kh,
    const unsigned char* __restrict__ mask, const h16* __restrict__ Vt,
    float* __restrict__ attn_out, h16* __restrict__ ctx) {
  extern __shared__ char smem_raw[];
  float* sc    = (float*)smem_raw;            // 32 x SCP f32 scores
  float* red   = sc + 32 * SCP;               // 128 f32 reduction scratch
  h16*   qtile = (h16*)(red + 128);           // 32 x 64 f16 Q tile

  const int t = threadIdx.x;
  const int lane = t & 31, l = lane & 15, hh = lane >> 4;
  const int w = t >> 5;
  const int z = blockIdx.z;                   // b*NH + h
  const int bb = z / NH, hid = z % NH;
  const int q0 = blockIdx.x * 32;

  const h16* Qz = Qh + ((size_t)z * SS + q0) * DK;   // 32x64, contiguous 4KB
  const h16* Kt = Kh + (size_t)z * SS * DK;          // [S x 64]
  const h16* Vz = Vt + (size_t)z * DK * SS;          // [64 x S]

  // ---- phase 0: Q tile -> LDS ----
#if defined(ASYNC_OK)
#pragma unroll
  for (int i = t; i < 256; i += 128)   // 256 * 16B = 4KB
    __builtin_amdgcn_global_load_async_to_lds_b128(
        (gv4i_p)(void*)(Qz + i * 8), (lv4i_p)(void*)(qtile + i * 8), 0, 0);
#if __has_builtin(__builtin_amdgcn_s_wait_asynccnt)
  __builtin_amdgcn_s_wait_asynccnt(0);
#else
  asm volatile("s_wait_asynccnt 0" ::: "memory");
#endif
#else
#pragma unroll
  for (int i = t; i < 256; i += 128)
    *reinterpret_cast<float4*>(qtile + i * 8) =
        *reinterpret_cast<const float4*>(Qz + i * 8);
#endif
  __syncthreads();

  // ---- phase 1: scores (raw, unmasked) into LDS ----
  v16h aq[2][2];
#pragma unroll
  for (int ti = 0; ti < 2; ++ti)
#pragma unroll
    for (int ks = 0; ks < 2; ++ks)
      aq[ti][ks] = frag_a_h(qtile, DK, ti * 16, ks * 32, l, hh);

#pragma unroll
  for (int nt = 0; nt < 16; ++nt) {
    const int n0 = w * 256 + nt * 16;
    v8f acc0 = {}, acc1 = {};
#pragma unroll
    for (int ks = 0; ks < 2; ++ks) {
      v16h bk = frag_b_h(Kt, DK, n0, ks * 32, l, hh);
      acc0 = wmma_f16(aq[0][ks], bk, acc0);
      acc1 = wmma_f16(aq[1][ks], bk, acc1);
    }
#pragma unroll
    for (int r = 0; r < 8; ++r) {
      sc[(r + hh * 8) * SCP + n0 + l]      = acc0[r];
      sc[(16 + r + hh * 8) * SCP + n0 + l] = acc1[r];
    }
  }
  __syncthreads();

  // ---- phase 2: mask + softmax (row = t&31, 4 chunks of 256 per row) ----
  {
    const int row = t & 31, part = t >> 5;
    float* r = sc + row * SCP + part * 256;
    const unsigned char* mrow =
        mask + ((size_t)bb * SS + q0 + row) * SS + part * 256;
    float mx = -3.402823466e+38f;
    for (int i = 0; i < 256; ++i) {
      float v = r[i];
      if (mrow[i]) v = -1e9f;
      r[i] = v;
      mx = fmaxf(mx, v);
    }
    red[t] = mx; __syncthreads();
    if (t < 32) {
      float m = fmaxf(fmaxf(red[t], red[t + 32]), fmaxf(red[t + 64], red[t + 96]));
      red[t] = m;
    }
    __syncthreads();
    const float rowmax = red[row];
    __syncthreads();           // red reused below
    float s = 0.f;
    for (int i = 0; i < 256; ++i) {
      float e = __expf(r[i] - rowmax);
      r[i] = e;
      s += e;
    }
    red[t] = s; __syncthreads();
    if (t < 32) red[t] = red[t] + red[t + 32] + red[t + 64] + red[t + 96];
    __syncthreads();
    const float inv = 1.0f / red[row];
    for (int i = 0; i < 256; ++i) r[i] *= inv;
  }
  __syncthreads();

  // ---- phase 2b: coalesced f32 probability write to d_out ----
  {
    float* out = attn_out + ((size_t)z * SS + q0) * SS;
    for (int i = 0; i < 64; ++i) {           // 32 rows * 256 float4 / 128 thr
      int idx = t + i * 128;
      int row = idx >> 8, c4 = idx & 255;
      float4 v = *reinterpret_cast<const float4*>(sc + row * SCP + c4 * 4);
      *reinterpret_cast<float4*>(out + (size_t)row * SS + c4 * 4) = v;
    }
  }

  // ---- phase 3: context = probs @ V ----
  {
    const int n0 = w * 16;                   // 4 waves x 16 cols = 64
    v8f c0 = {}, c1 = {};
    for (int k0 = 0; k0 < SS; k0 += 32) {
      v16h a0 = frag_a_f32(sc, SCP, 0, k0, l, hh);
      v16h a1 = frag_a_f32(sc, SCP, 16, k0, l, hh);
      v16h bv = frag_b_h(Vz, SS, n0, k0, l, hh);
      c0 = wmma_f16(a0, bv, c0);
      c1 = wmma_f16(a1, bv, c1);
    }
#pragma unroll
    for (int r = 0; r < 8; ++r) {
      const int d = n0 + l;
      const int qa = q0 + r + hh * 8;
      const int qb = qa + 16;
      ctx[((size_t)bb * SS + qa) * D_MODEL + hid * DK + d] = (h16)c0[r];
      ctx[((size_t)bb * SS + qb) * D_MODEL + hid * DK + d] = (h16)c1[r];
    }
  }
}

// ---------------- residual add + LayerNorm -> f16 ----------------
__global__ void __launch_bounds__(256) residual_ln(
    const float* __restrict__ wo_out, const float* __restrict__ x,
    const float* __restrict__ gamma, const float* __restrict__ beta,
    h16* __restrict__ out_h) {
  const int row = blockIdx.x;
  const float* a = wo_out + (size_t)row * D_MODEL;
  const float* b = x      + (size_t)row * D_MODEL;
  __shared__ float red[256];
  const int t = threadIdx.x;

  float v[3], s = 0.f;
#pragma unroll
  for (int i = 0; i < 3; ++i) { v[i] = a[t + i * 256] + b[t + i * 256]; s += v[i]; }
  red[t] = s; __syncthreads();
  for (int o = 128; o > 0; o >>= 1) { if (t < o) red[t] += red[t + o]; __syncthreads(); }
  const float mean = red[0] * (1.0f / D_MODEL);
  __syncthreads();

  float q = 0.f;
#pragma unroll
  for (int i = 0; i < 3; ++i) { float d = v[i] - mean; q += d * d; }
  red[t] = q; __syncthreads();
  for (int o = 128; o > 0; o >>= 1) { if (t < o) red[t] += red[t + o]; __syncthreads(); }
  const float rstd = rsqrtf(red[0] * (1.0f / D_MODEL) + 1e-5f);

#pragma unroll
  for (int i = 0; i < 3; ++i) {
    const int c = t + i * 256;
    out_h[(size_t)row * D_MODEL + c] = (h16)((v[i] - mean) * rstd * gamma[c] + beta[c]);
  }
}

// ---------------------------------------------------------------------------
extern "C" void kernel_launch(void* const* d_in, const int* in_sizes, int n_in,
                              void* d_out, int out_size, void* d_ws, size_t ws_size,
                              hipStream_t stream) {
  const float* x     = (const float*)d_in[0];
  const unsigned char* mask = (const unsigned char*)d_in[1];  // jax bool = 1 byte
  const float* W_q = (const float*)d_in[2];  const float* b_q = (const float*)d_in[3];
  const float* W_k = (const float*)d_in[4];  const float* b_k = (const float*)d_in[5];
  const float* W_v = (const float*)d_in[6];  const float* b_v = (const float*)d_in[7];
  const float* W_o = (const float*)d_in[8];  const float* b_o = (const float*)d_in[9];
  const float* ln_g = (const float*)d_in[10]; const float* ln_b = (const float*)d_in[11];
  const float* W1  = (const float*)d_in[12]; const float* b1  = (const float*)d_in[13];
  const float* W2  = (const float*)d_in[14]; const float* b2  = (const float*)d_in[15];

  float* ffn_out  = (float*)d_out;                               // [8192 x 768]
  float* attn_out = (float*)d_out + (size_t)MTOT * D_MODEL;      // [96x1024x1024]

  char* ws = (char*)d_ws;
  size_t off = 0;
  auto carve = [&](size_t bytes) { char* p = ws + off; off += (bytes + 255) & ~(size_t)255; return p; };
  h16* x_h   = (h16*)carve((size_t)MTOT * D_MODEL * 2);
  h16* Wqt   = (h16*)carve((size_t)D_MODEL * D_MODEL * 2);
  h16* Wkt   = (h16*)carve((size_t)D_MODEL * D_MODEL * 2);
  h16* Wvt   = (h16*)carve((size_t)D_MODEL * D_MODEL * 2);
  h16* Wot   = (h16*)carve((size_t)D_MODEL * D_MODEL * 2);
  h16* W1t   = (h16*)carve((size_t)DFF * D_MODEL * 2);
  h16* W2t   = (h16*)carve((size_t)D_MODEL * DFF * 2);
  h16* Q_h   = (h16*)carve((size_t)BB * NH * SS * DK * 2);
  h16* K_h   = (h16*)carve((size_t)BB * NH * SS * DK * 2);
  h16* Vt_h  = (h16*)carve((size_t)BB * NH * DK * SS * 2);
  h16* ctx_h = (h16*)carve((size_t)MTOT * D_MODEL * 2);
  float* wo_f = (float*)carve((size_t)MTOT * D_MODEL * 4);
  h16* ao_h  = (h16*)carve((size_t)MTOT * D_MODEL * 2);
  h16* h1_h  = (h16*)carve((size_t)MTOT * DFF * 2);
  (void)ws_size; (void)n_in; (void)in_sizes; (void)out_size;

  // 1) conversions
  {
    int n = MTOT * D_MODEL;
    cvt_f32_to_f16<<<(n + 255) / 256, 256, 0, stream>>>(x, x_h, n);
    int nw = D_MODEL * D_MODEL;
    cvt_transpose_h<<<(nw + 255) / 256, 256, 0, stream>>>(W_q, Wqt, D_MODEL, D_MODEL);
    cvt_transpose_h<<<(nw + 255) / 256, 256, 0, stream>>>(W_k, Wkt, D_MODEL, D_MODEL);
    cvt_transpose_h<<<(nw + 255) / 256, 256, 0, stream>>>(W_v, Wvt, D_MODEL, D_MODEL);
    cvt_transpose_h<<<(nw + 255) / 256, 256, 0, stream>>>(W_o, Wot, D_MODEL, D_MODEL);
    int nf = D_MODEL * DFF;
    cvt_transpose_h<<<(nf + 255) / 256, 256, 0, stream>>>(W1, W1t, D_MODEL, DFF);
    cvt_transpose_h<<<(nf + 255) / 256, 256, 0, stream>>>(W2, W2t, DFF, D_MODEL);
  }

  // 2) QKV projections (Q pre-scaled by 1/sqrt(dk))
  {
    dim3 grid(D_MODEL / 128, MTOT / 128);
    gemm_ws<<<grid, 128, 0, stream>>>(x_h, Wqt, b_q, D_MODEL, D_MODEL, D_MODEL,
                                      nullptr, Q_h, 0, EPI_QK, 0.125f);
    gemm_ws<<<grid, 128, 0, stream>>>(x_h, Wkt, b_k, D_MODEL, D_MODEL, D_MODEL,
                                      nullptr, K_h, 0, EPI_QK, 1.0f);
    gemm_ws<<<grid, 128, 0, stream>>>(x_h, Wvt, b_v, D_MODEL, D_MODEL, D_MODEL,
                                      nullptr, Vt_h, 0, EPI_VT, 1.0f);
  }

  // 3) fused attention: scores + mask + softmax (-> d_out attn) + context
  {
    const int smem = 32 * SCP * 4 + 128 * 4 + 32 * DK * 2;  // 136192 B
    (void)hipFuncSetAttribute(reinterpret_cast<const void*>(fused_attn),
                              hipFuncAttributeMaxDynamicSharedMemorySize, smem);
    fused_attn<<<dim3(SS / 32, 1, BB * NH), 128, smem, stream>>>(
        Q_h, K_h, mask, Vt_h, attn_out, ctx_h);
  }

  // 4) output projection -> f32, then residual + LayerNorm -> f16
  gemm_ws<<<dim3(D_MODEL / 128, MTOT / 128), 128, 0, stream>>>(
      ctx_h, Wot, b_o, D_MODEL, D_MODEL, D_MODEL, wo_f, nullptr, D_MODEL, EPI_F32, 1.0f);
  residual_ln<<<MTOT, 256, 0, stream>>>(wo_f, x, ln_g, ln_b, ao_h);

  // 5) FFN: gelu(ao @ W1 + b1) @ W2 + b2 -> d_out[0..]
  gemm_ws<<<dim3(DFF / 128, MTOT / 128), 128, 0, stream>>>(
      ao_h, W1t, b1, D_MODEL, D_MODEL, D_MODEL, nullptr, h1_h, DFF, EPI_GELU_F16, 1.0f);
  gemm_ws<<<dim3(D_MODEL / 128, MTOT / 128), 128, 0, stream>>>(
      h1_h, W2t, b2, DFF, DFF, DFF, ffn_out, nullptr, D_MODEL, EPI_F32, 1.0f);
}
